// NodeModelAggrByEdge_62766652064043
// MI455X (gfx1250) — matrix operand order repaired and checked by reading, not compiled, not run on hardware
//
#include <hip/hip_runtime.h>
#include <hip/hip_bf16.h>

// Problem constants (match reference)
#define NN   100000   // nodes
#define EE   800000   // edges per relation
#define TT   3        // relations
#define DD   128      // feature dim (in == out)

typedef float    v2f  __attribute__((ext_vector_type(2)));
typedef float    v8f  __attribute__((ext_vector_type(8)));
typedef _Float16 v16h __attribute__((ext_vector_type(16)));
typedef unsigned int u32x4 __attribute__((ext_vector_type(4)));
typedef int          i32x4 __attribute__((ext_vector_type(4)));
typedef int          i32x8 __attribute__((ext_vector_type(8)));

#if __has_builtin(__builtin_amdgcn_wmma_f32_16x16x4_f32)
#define USE_F32_WMMA 1
#else
#define USE_F32_WMMA 0
#endif

#if defined(__gfx1250__) && __has_builtin(__builtin_amdgcn_tensor_load_to_lds) && \
    __has_builtin(__builtin_amdgcn_s_wait_tensorcnt)
#define HAVE_TDM 1
#else
#define HAVE_TDM 0
#endif

#if __has_include(<hip/amd_detail/amd_gfx1250_TDM.h>)
#define TDM_6ARG 1
#else
#define TDM_6ARG 0
#endif

#define MB  64   // M rows per block (4 x 16 tiles per wave)
#define LDA 132  // padded LDS row stride in floats (132 % 64 = 4 -> no bank conflicts)

// ---------------------------------------------------------------------------
// xw[m0:m0+64, :] = x[m0:m0+64, :] @ W   (one relation)
// 8 waves; wave w owns output cols [16w,16w+16); 4 M-tiles per wave so each
// B load pair feeds 4 WMMAs. A tile staged to LDS by the Tensor Data Mover
// (hardware pad -> stride 132) when available, else manual staging.
// ---------------------------------------------------------------------------
__global__ __launch_bounds__(256) void gemm_xw_kernel(
    const float* __restrict__ x, const float* __restrict__ W,
    float* __restrict__ xw) {
  __shared__ float lA[MB * LDA];
  const int tid  = threadIdx.x;
  const int m0   = blockIdx.x * MB;
  const int rows = (NN - m0 < MB) ? (NN - m0) : MB;

#if HAVE_TDM
  if (tid < 32) {  // one wave issues the DMA (EXEC ignored by TDM)
    const unsigned lds_base =
        (unsigned)(unsigned long long)(uintptr_t)&lA[0];
    const unsigned long long ga =
        (unsigned long long)(uintptr_t)(x + (size_t)m0 * DD);
    // D# group 0: count=1, lds_addr, 57-bit global_addr, type=2 ("image")
    u32x4 g0;
    g0[0] = 1u;
    g0[1] = lds_base;
    g0[2] = (unsigned)(ga & 0xFFFFFFFFu);
    g0[3] = (unsigned)((ga >> 32) & 0x1FFFFFFu) | (2u << 30);
    // D# group 1: data_size=4B(code2), pad_enable, pad_interval=128DW(code6),
    // pad_amount=4DW(code3); tensor 128 x N, tile 128 x rows, stride 128.
    i32x8 g1;
    g1[0] = (2 << 16) | (1 << 20) | (6 << 22) | (3 << 25);
    g1[1] = (int)(128u << 16);                       // tensor_dim0[15:0] @ [63:48]
    g1[2] = (int)(((unsigned)NN & 0xFFFFu) << 16);   // dim0 hi=0 | tensor_dim1 lo
    g1[3] = (int)(((unsigned)NN >> 16) | (128u << 16)); // dim1 hi | tile_dim0
    g1[4] = rows;                                    // tile_dim1 (tile_dim2=0)
    g1[5] = 128;                                     // tensor_dim0_stride lo32
    g1[6] = 0;
    g1[7] = 0;
    const i32x4 gz4 = {0, 0, 0, 0};
#if TDM_6ARG
    const i32x8 gz8 = {0, 0, 0, 0, 0, 0, 0, 0};
    __builtin_amdgcn_tensor_load_to_lds(g0, g1, gz4, gz4, gz8, 0);
#else
    __builtin_amdgcn_tensor_load_to_lds(g0, g1, gz4, gz4, 0);
#endif
    __builtin_amdgcn_s_wait_tensorcnt(0);
  }
  __syncthreads();
#else
  for (int i = tid; i < MB * DD; i += 256) {
    const int r = i >> 7, c = i & 127;
    const int rr = (m0 + r < NN) ? (m0 + r) : (NN - 1);
    lA[r * LDA + c] = x[(size_t)rr * DD + c];
  }
  __syncthreads();
#endif

  const int w  = tid >> 5;        // wave id -> output cols [16w, 16w+16)
  const int l  = tid & 31;
  const int m  = l & 15;
  const int hi = l >> 4;
  const int n  = w * 16 + m;

  v8f acc0 = {}, acc1 = {}, acc2 = {}, acc3 = {};
#if USE_F32_WMMA
#pragma unroll
  for (int kk = 0; kk < DD; kk += 4) {
    const int ka = kk + 2 * hi;     // A/B: VGPR j holds K = 2*hi + j
    v2f b;
    b.x = W[(size_t)ka * DD + n];
    b.y = W[(size_t)(ka + 1) * DD + n];
    v2f a0, a1, a2, a3;
    a0.x = lA[(m     ) * LDA + ka]; a0.y = lA[(m     ) * LDA + ka + 1];
    a1.x = lA[(m + 16) * LDA + ka]; a1.y = lA[(m + 16) * LDA + ka + 1];
    a2.x = lA[(m + 32) * LDA + ka]; a2.y = lA[(m + 32) * LDA + ka + 1];
    a3.x = lA[(m + 48) * LDA + ka]; a3.y = lA[(m + 48) * LDA + ka + 1];
    acc0 = __builtin_amdgcn_wmma_f32_16x16x4_f32(false, a0, false, b, (short)0, acc0, false, false);
    acc1 = __builtin_amdgcn_wmma_f32_16x16x4_f32(false, a1, false, b, (short)0, acc1, false, false);
    acc2 = __builtin_amdgcn_wmma_f32_16x16x4_f32(false, a2, false, b, (short)0, acc2, false, false);
    acc3 = __builtin_amdgcn_wmma_f32_16x16x4_f32(false, a3, false, b, (short)0, acc3, false, false);
  }
#else
#pragma unroll
  for (int kk = 0; kk < DD; kk += 32) {
    v16h b;
#pragma unroll
    for (int e = 0; e < 16; ++e)
      b[e] = (_Float16)W[(size_t)(kk + 16 * hi + e) * DD + n];
    v16h a0, a1, a2, a3;
#pragma unroll
    for (int e = 0; e < 16; ++e) {
      const int ko = 16 * hi + e;
      a0[e] = (_Float16)lA[(m     ) * LDA + kk + ko];
      a1[e] = (_Float16)lA[(m + 16) * LDA + kk + ko];
      a2[e] = (_Float16)lA[(m + 32) * LDA + kk + ko];
      a3[e] = (_Float16)lA[(m + 48) * LDA + kk + ko];
    }
    acc0 = __builtin_amdgcn_wmma_f32_16x16x32_f16(false, a0, false, b, (short)0, acc0, false, false);
    acc1 = __builtin_amdgcn_wmma_f32_16x16x32_f16(false, a1, false, b, (short)0, acc1, false, false);
    acc2 = __builtin_amdgcn_wmma_f32_16x16x32_f16(false, a2, false, b, (short)0, acc2, false, false);
    acc3 = __builtin_amdgcn_wmma_f32_16x16x32_f16(false, a3, false, b, (short)0, acc3, false, false);
  }
#endif

#pragma unroll
  for (int r = 0; r < 8; ++r) {
    const int rb = r + 8 * hi;        // D: VGPR r -> tile row r + 8*hi
    if (m0 + rb      < NN) xw[(size_t)(m0 + rb     ) * DD + n] = acc0[r];
    if (m0 + 16 + rb < NN) xw[(size_t)(m0 + 16 + rb) * DD + n] = acc1[r];
    if (m0 + 32 + rb < NN) xw[(size_t)(m0 + 32 + rb) * DD + n] = acc2[r];
    if (m0 + 48 + rb < NN) xw[(size_t)(m0 + 48 + rb) * DD + n] = acc3[r];
  }
}

// ---------------------------------------------------------------------------
// Degree / normalization
// ---------------------------------------------------------------------------
__global__ __launch_bounds__(256) void init_deg_kernel(unsigned int* deg) {
  const int i = blockIdx.x * 256 + threadIdx.x;
  if (i < TT * NN) deg[i] = 1u;  // self-loop
}

__global__ __launch_bounds__(256) void accum_deg_kernel(
    const int* __restrict__ edges, unsigned int* __restrict__ deg) {
  const int i = blockIdx.x * 256 + threadIdx.x;  // over T*E
  if (i >= TT * EE) return;
  const int t = i / EE;
  const int e = i - t * EE;
  const int d = edges[(size_t)t * 2 * EE + EE + e];  // dst row
  __hip_atomic_fetch_add(&deg[t * NN + d], 1u,
                         __ATOMIC_RELAXED, __HIP_MEMORY_SCOPE_AGENT);
}

__global__ __launch_bounds__(256) void calc_dis_kernel(
    const unsigned int* __restrict__ deg, float* __restrict__ dis) {
  const int i = blockIdx.x * 256 + threadIdx.x;
  if (i < TT * NN) dis[i] = rsqrtf((float)deg[i]);
}

// ---------------------------------------------------------------------------
// Self-loop contribution + bias; initializes out at t==0. Race-free: each
// (node, col4) owned by exactly one thread.
// ---------------------------------------------------------------------------
__global__ __launch_bounds__(256) void self_bias_kernel(
    const float* __restrict__ xw, const float* __restrict__ dis,
    const float* __restrict__ bias, float* __restrict__ out, int initOut) {
  const int i = blockIdx.x * 256 + threadIdx.x;  // over N*32 (exact multiple)
  const int v = i >> 5;
  const int c = (i & 31) * 4;
  const float d = dis[v];
  const float s = d * d;
  const float4 xv = *(const float4*)(xw + (size_t)v * DD + c);
  float4 acc;
  if (initOut)
    acc = make_float4(bias[c], bias[c + 1], bias[c + 2], bias[c + 3]);
  else
    acc = *(const float4*)(out + (size_t)v * DD + c);
  acc.x += xv.x * s; acc.y += xv.y * s;
  acc.z += xv.z * s; acc.w += xv.w * s;
  *(float4*)(out + (size_t)v * DD + c) = acc;
}

// ---------------------------------------------------------------------------
// Edge scatter: 32 lanes per edge, float4 per lane, native f32 atomics.
// ---------------------------------------------------------------------------
__global__ __launch_bounds__(256) void scatter_kernel(
    const int* __restrict__ ei,        // [2, E] for this relation
    const float* __restrict__ xw, const float* __restrict__ dis,
    float* __restrict__ out) {
  const int e    = blockIdx.x * 8 + (threadIdx.x >> 5);  // E % 8 == 0
  const int lane = threadIdx.x & 31;
  const int s = ei[e];
  const int d = ei[EE + e];
  const float norm = dis[s] * dis[d];
  const int c = lane * 4;
  const float4 xv = *(const float4*)(xw + (size_t)s * DD + c);
  float* o = out + (size_t)d * DD + c;
  __hip_atomic_fetch_add(o + 0, xv.x * norm, __ATOMIC_RELAXED, __HIP_MEMORY_SCOPE_AGENT);
  __hip_atomic_fetch_add(o + 1, xv.y * norm, __ATOMIC_RELAXED, __HIP_MEMORY_SCOPE_AGENT);
  __hip_atomic_fetch_add(o + 2, xv.z * norm, __ATOMIC_RELAXED, __HIP_MEMORY_SCOPE_AGENT);
  __hip_atomic_fetch_add(o + 3, xv.w * norm, __ATOMIC_RELAXED, __HIP_MEMORY_SCOPE_AGENT);
}

// ---------------------------------------------------------------------------
extern "C" void kernel_launch(void* const* d_in, const int* in_sizes, int n_in,
                              void* d_out, int out_size, void* d_ws, size_t ws_size,
                              hipStream_t stream) {
  const float* x     = (const float*)d_in[0];   // [N, 128]
  const int*   edges = (const int*)d_in[1];     // [T, 2, E]
  const float* W     = (const float*)d_in[2];   // [T, 128, 128]
  const float* b     = (const float*)d_in[3];   // [T, 128]
  float*       out   = (float*)d_out;           // [N, 128]

  // Workspace layout: xw | deg | dis
  char* ws = (char*)d_ws;
  float*        xw  = (float*)ws;                                   // N*128 f32
  unsigned int* deg = (unsigned int*)(ws + (size_t)NN * DD * 4);    // T*N u32
  float*        dis = (float*)(ws + (size_t)NN * DD * 4
                                  + (size_t)TT * NN * 4);           // T*N f32

  // 1) degrees (self-loop + incoming edges), then D^-1/2
  init_deg_kernel<<<(TT * NN + 255) / 256, 256, 0, stream>>>(deg);
  accum_deg_kernel<<<(TT * EE + 255) / 256, 256, 0, stream>>>(edges, deg);
  calc_dis_kernel<<<(TT * NN + 255) / 256, 256, 0, stream>>>(deg, dis);

  // 2) per-relation: GEMM (WMMA + TDM staging) -> self-loop+bias -> scatter
  for (int t = 0; t < TT; ++t) {
    gemm_xw_kernel<<<(NN + MB - 1) / MB, 256, 0, stream>>>(
        x, W + (size_t)t * DD * DD, xw);
    self_bias_kernel<<<(NN * 32) / 256, 256, 0, stream>>>(
        xw, dis + (size_t)t * NN, b + (size_t)t * DD, out, t == 0 ? 1 : 0);
    scatter_kernel<<<EE / 8, 256, 0, stream>>>(
        edges + (size_t)t * 2 * EE, xw, dis + (size_t)t * NN, out);
  }
}
